// ScalarPotentialLMSARF_48180943127153
// MI455X (gfx1250) — matrix operand order repaired and checked by reading, not compiled, not run on hardware
//
#include <hip/hip_runtime.h>

typedef __attribute__((ext_vector_type(16))) __bf16        v16bf;
typedef __attribute__((ext_vector_type(8)))  float         v8f;
typedef __attribute__((ext_vector_type(4)))  unsigned int  u32x4;
typedef __attribute__((ext_vector_type(4)))  unsigned int  uint32x4_t;
typedef __attribute__((ext_vector_type(8)))  int           int32x8_t;
typedef __attribute__((ext_vector_type(4)))  int           int32x4_t;

#ifndef __has_builtin
#define __has_builtin(x) 0
#endif
#if __has_builtin(__builtin_amdgcn_tensor_load_to_lds)
#define HAVE_TDM 1
#else
#define HAVE_TDM 0
#endif

#define NTOK    4096
#define DMODEL  128
#define HID_N   512
#define VOCAB_N 50257
#define TSEQ    512
#define NLAYER  8

#define LDAP 520   // padded row stride (bf16 elems) for 512-wide LDS tiles
#define ZLDA 264   // padded row stride for 256-wide z tile
#define HLDA 136   // padded row stride for 128-wide h tile

// ---------------- workspace layout (bytes) ----------------
static constexpr size_t OFF_H   = 0;
static constexpr size_t OFF_V   = OFF_H   + (size_t)NTOK * DMODEL * 4;
static constexpr size_t OFF_XI  = OFF_V   + (size_t)NTOK * DMODEL * 4;
static constexpr size_t OFF_W0B = OFF_XI  + (size_t)NTOK * DMODEL * 4;   // W0 bf16 row-major (256x512)
static constexpr size_t OFF_W0T = OFF_W0B + (size_t)256 * 512 * 2;       // W0^T bf16 (512x256)
static constexpr size_t OFF_W1B = OFF_W0T + (size_t)512 * 256 * 2;
static constexpr size_t OFF_W1T = OFF_W1B + (size_t)512 * 512 * 2;
static constexpr size_t OFF_W2B = OFF_W1T + (size_t)512 * 512 * 2;
static constexpr size_t OFF_W2T = OFF_W2B + (size_t)512 * 512 * 2;
static constexpr size_t OFF_EB  = OFF_W2T + (size_t)512 * 512 * 2;       // E bf16 (50257x128)
static constexpr size_t OFF_HB  = OFF_EB  + (size_t)VOCAB_N * DMODEL * 2;// h bf16 (4096x128)

static constexpr int MLP_LDS_BYTES = (32 * ZLDA + 4 * 32 * LDAP) * 2;    // 150016 B

// ---------------- small helpers ----------------
__device__ __forceinline__ unsigned short f2bf(float f) {
  unsigned int u = __float_as_uint(f);
  u += 0x7FFFu + ((u >> 16) & 1u);          // round-to-nearest-even
  return (unsigned short)(u >> 16);
}
__device__ __forceinline__ float bf2f(unsigned short s) {
  return __uint_as_float(((unsigned int)s) << 16);
}
__device__ __forceinline__ float gelu_f(float x) {
  return 0.5f * x * (1.0f + erff(x * 0.7071067811865476f));
}
__device__ __forceinline__ float gelu_g(float x) {
  return 0.5f * (1.0f + erff(x * 0.7071067811865476f))
       + x * 0.3989422804014327f * __expf(-0.5f * x * x);
}
__device__ __forceinline__ float softplus_f(float x) {
  return (x > 20.0f) ? x : log1pf(__expf(x));
}
__device__ __forceinline__ v16bf ld_bf16x16(const unsigned short* p0, const unsigned short* p1) {
  union { u32x4 q[2]; v16bf v; } u;
  u.q[0] = *(const u32x4*)p0;
  u.q[1] = *(const u32x4*)p1;
  return u.v;
}
__device__ __forceinline__ v16bf bf_zero() {
  union { unsigned int w[8]; v16bf v; } u;
#pragma unroll
  for (int i = 0; i < 8; ++i) u.w[i] = 0u;
  return u.v;
}

// NACC output tiles sharing one A stripe (rows mrow..mrow+15), columns ncol0 + i*64.
// Bcm is N-major: Bcm[n*ldb + k].  A per ISA 7.12.2 16-bit A 16x32 layout, B 32x16.
template <int NACC>
__device__ __forceinline__ void bf16_gemm_multi(
    v8f (&acc)[NACC], const unsigned short* A, int lda, int mrow,
    const unsigned short* Bcm, int ldb, int ncol0, int K, int lane) {
  const int half = lane >> 4;
  const int r    = lane & 15;
  const unsigned short* pa  = A   + (size_t)(mrow + r) * lda + half * 8;
  const unsigned short* pb0 = Bcm + (size_t)(ncol0 + r) * ldb + half * 8;
  for (int kb = 0; kb < K; kb += 32) {
    v16bf a = ld_bf16x16(pa + kb, pa + kb + 16);
#pragma unroll
    for (int i = 0; i < NACC; ++i) {
      const unsigned short* pb = pb0 + (size_t)(i * 64) * ldb + kb;
      v16bf b = ld_bf16x16(pb, pb + 16);
      acc[i] = __builtin_amdgcn_wmma_f32_16x16x32_bf16(false, a, false, b, (short)0, acc[i], false, false);
    }
  }
}

// ---------------- prep kernels ----------------
__global__ void conv_w_kernel(const float* __restrict__ W,
                              unsigned short* __restrict__ Wb,
                              unsigned short* __restrict__ WTb, int K, int N) {
  int idx = blockIdx.x * blockDim.x + threadIdx.x;
  if (idx >= K * N) return;
  int k = idx / N, n = idx - k * N;
  unsigned short bv = f2bf(W[idx]);
  Wb[idx] = bv;
  WTb[(size_t)n * K + k] = bv;
}

__global__ void conv_e_kernel(const float* __restrict__ E,
                              unsigned short* __restrict__ Eb, int n) {
  int idx = blockIdx.x * blockDim.x + threadIdx.x;
  if (idx < n) Eb[idx] = f2bf(E[idx]);
}

__global__ void embed_kernel(const int* __restrict__ x, const float* __restrict__ E,
                             const float* __restrict__ P,
                             float* __restrict__ h, float* __restrict__ v) {
  int idx = blockIdx.x * blockDim.x + threadIdx.x;
  if (idx >= NTOK * DMODEL) return;
  int tok = idx >> 7, d = idx & 127;
  int t = tok & (TSEQ - 1);
  h[idx] = E[(size_t)x[tok] * DMODEL + d] + P[(size_t)t * DMODEL + d];
  v[idx] = 0.0f;
}

// xi[b,t,:] = mean_{s<=t} h[b,s,:]
__global__ void cumsum_kernel(const float* __restrict__ h, float* __restrict__ xi) {
  int b = blockIdx.x;
  int d = threadIdx.x;
  const float* hp = h  + (size_t)b * TSEQ * DMODEL + d;
  float*       xp = xi + (size_t)b * TSEQ * DMODEL + d;
  float run = 0.0f;
  for (int t = 0; t < TSEQ; ++t) {
    run += hp[(size_t)t * DMODEL];
    xp[(size_t)t * DMODEL] = run / (float)(t + 1);
  }
}

// ---------------- fused MLP fwd+bwd + integrator (one layer step) ----------------
__global__ __launch_bounds__(256) void sarf_mlp_fused(
    const float* __restrict__ xi, float* __restrict__ h, float* __restrict__ v,
    unsigned short* __restrict__ hbg,
    const unsigned short* __restrict__ W0T, const unsigned short* __restrict__ W0B,
    const unsigned short* __restrict__ W1T, const unsigned short* __restrict__ W1B,
    const unsigned short* __restrict__ W2T, const unsigned short* __restrict__ W2B,
    const float* __restrict__ b0, const float* __restrict__ b1,
    const float* __restrict__ b2, const float* __restrict__ W3,
    const float* __restrict__ raw_m, const float* __restrict__ raw_gamma) {
  extern __shared__ unsigned short smem[];
  unsigned short* zb   = smem;                 // 32 x ZLDA : concat(xi,h) bf16
  unsigned short* bufA = zb   + 32 * ZLDA;     // 32 x LDAP : z1 -> dA2 -> dA0
  unsigned short* gp0  = bufA + 32 * LDAP;     // 32 x LDAP : gelu'(a0)
  unsigned short* bufB = gp0  + 32 * LDAP;     // 32 x LDAP : z2 -> dA1
  unsigned short* gp1  = bufB + 32 * LDAP;     // 32 x LDAP : gelu'(a1)

  const int tid  = threadIdx.x;
  const int tok0 = blockIdx.x * 32;
  const int wave = tid >> 5;
  const int lane = tid & 31;
  const int half = lane >> 4;
  const int r    = lane & 15;
  const int mrow  = (wave & 1) * 16;
  const int ncol0 = (wave >> 1) * 16;

  // build z = [xi, h] tile in bf16 (padded rows)
  for (int i = tid; i < 32 * 256; i += 256) {
    const int m = i >> 8, k = i & 255;
    const int t = tok0 + m;
    const float val = (k < DMODEL) ? xi[(size_t)t * DMODEL + k]
                                   : h[(size_t)t * DMODEL + (k - DMODEL)];
    zb[m * ZLDA + k] = f2bf(val);
  }
  __syncthreads();

  // Stage 1: a0 = z @ W0 + b0 ; z1 = gelu(a0) -> bufA ; gp0 = gelu'(a0)
  {
    v8f acc[8] = {};
    bf16_gemm_multi<8>(acc, zb, ZLDA, mrow, W0T, 256, ncol0, 256, lane);
#pragma unroll
    for (int i = 0; i < 8; ++i) {
      const int ncol = ncol0 + i * 64;
      const float bv = b0[ncol + r];
#pragma unroll
      for (int j = 0; j < 8; ++j) {
        const float a = acc[i][j] + bv;
        const int idx = (mrow + half * 8 + j) * LDAP + (ncol + r);
        bufA[idx] = f2bf(gelu_f(a));
        gp0[idx]  = f2bf(gelu_g(a));
      }
    }
  }
  __syncthreads();

  // Stage 2: a1 = z1 @ W1 + b1 ; z2 -> bufB ; gp1 = gelu'(a1)
  {
    v8f acc[8] = {};
    bf16_gemm_multi<8>(acc, bufA, LDAP, mrow, W1T, 512, ncol0, 512, lane);
#pragma unroll
    for (int i = 0; i < 8; ++i) {
      const int ncol = ncol0 + i * 64;
      const float bv = b1[ncol + r];
#pragma unroll
      for (int j = 0; j < 8; ++j) {
        const float a = acc[i][j] + bv;
        const int idx = (mrow + half * 8 + j) * LDAP + (ncol + r);
        bufB[idx] = f2bf(gelu_f(a));
        gp1[idx]  = f2bf(gelu_g(a));
      }
    }
  }
  __syncthreads();

  // Stage 3: a2 = z2 @ W2 + b2 ; dA2 = gelu'(a2) * W3[n] -> bufA
  {
    v8f acc[8] = {};
    bf16_gemm_multi<8>(acc, bufB, LDAP, mrow, W2T, 512, ncol0, 512, lane);
#pragma unroll
    for (int i = 0; i < 8; ++i) {
      const int ncol = ncol0 + i * 64;
      const float bv  = b2[ncol + r];
      const float w3v = W3[ncol + r];
#pragma unroll
      for (int j = 0; j < 8; ++j) {
        const int idx = (mrow + half * 8 + j) * LDAP + (ncol + r);
        bufA[idx] = f2bf(gelu_g(acc[i][j] + bv) * w3v);
      }
    }
  }
  __syncthreads();

  // Stage 4: dA1 = gp1 * (dA2 @ W2^T) -> bufB   (col-major of W2^T == W2 row-major)
  {
    v8f acc[8] = {};
    bf16_gemm_multi<8>(acc, bufA, LDAP, mrow, W2B, 512, ncol0, 512, lane);
#pragma unroll
    for (int i = 0; i < 8; ++i) {
      const int ncol = ncol0 + i * 64;
#pragma unroll
      for (int j = 0; j < 8; ++j) {
        const int idx = (mrow + half * 8 + j) * LDAP + (ncol + r);
        bufB[idx] = f2bf(acc[i][j] * bf2f(gp1[idx]));
      }
    }
  }
  __syncthreads();

  // Stage 5: dA0 = gp0 * (dA1 @ W1^T) -> bufA
  {
    v8f acc[8] = {};
    bf16_gemm_multi<8>(acc, bufB, LDAP, mrow, W1B, 512, ncol0, 512, lane);
#pragma unroll
    for (int i = 0; i < 8; ++i) {
      const int ncol = ncol0 + i * 64;
#pragma unroll
      for (int j = 0; j < 8; ++j) {
        const int idx = (mrow + half * 8 + j) * LDAP + (ncol + r);
        bufA[idx] = f2bf(acc[i][j] * bf2f(gp0[idx]));
      }
    }
  }
  __syncthreads();

  // Stage 6: grad_h = dA0 @ W0[128:256,:]^T ; v = (v - grad/m)/(1+gamma); h += v  (DT=1)
  {
    const float minv = 1.0f / (softplus_f(raw_m[0]) + 0.001f);
    const float dinv = 1.0f / (1.0f + softplus_f(raw_gamma[0]));
    v8f acc[2] = {};
    bf16_gemm_multi<2>(acc, bufA, LDAP, mrow, W0B + 128 * 512, 512, ncol0, 512, lane);
#pragma unroll
    for (int i = 0; i < 2; ++i) {
      const int ncol = ncol0 + i * 64;
#pragma unroll
      for (int j = 0; j < 8; ++j) {
        const int m  = mrow + half * 8 + j;
        const int gi = (tok0 + m) * DMODEL + (ncol + r);
        const float vv = (v[gi] - acc[i][j] * minv) * dinv;
        const float hn = h[gi] + vv;
        v[gi] = vv;
        h[gi] = hn;
        hbg[gi] = f2bf(hn);   // bf16 mirror for the LM head / TDM staging
      }
    }
  }
}

// ---------------- LM head: logits = h @ E^T ----------------
__global__ __launch_bounds__(256) void sarf_logits(
    const unsigned short* __restrict__ hbg, const unsigned short* __restrict__ Eb,
    float* __restrict__ out) {
  __shared__ unsigned short hb[32 * HLDA];
  const int tid  = threadIdx.x;
  const int tok0 = blockIdx.x * 32;
  const int v0   = blockIdx.y * 512;

#if HAVE_TDM
  // Tensor Data Mover: DMA the 32x128 bf16 h-tile into LDS with row padding
  // (pad_interval = 64 DWORDs -> after each 128-elem row, pad_amount = 4 DWORDs -> +8 elems)
  if (tid == 0) {
    const unsigned long long gaddr =
        (unsigned long long)(const void*)(hbg + (size_t)tok0 * DMODEL);
    const unsigned int lds_off = (unsigned int)(unsigned long long)(const void*)hb;
    uint32x4_t g0; int32x8_t g1; int32x4_t g2, g3;
    g0[0] = 1u;                                       // count=1 valid user descriptor
    g0[1] = lds_off;                                  // lds_addr
    g0[2] = (unsigned int)(gaddr & 0xFFFFFFFFull);    // global_addr[31:0]
    g0[3] = (unsigned int)((gaddr >> 32) & 0x01FFFFFFull) | 0x80000000u; // [56:32] | type=2
    g1[0] = (int)((1u << 16) | (1u << 20) | (5u << 22) | (3u << 25)); // data_size=2B, pad en, 64DW, 4DW
    g1[1] = (int)((unsigned)DMODEL << 16);            // tensor_dim0 = 128
    g1[2] = (int)(32u << 16);                         // tensor_dim1 = 32
    g1[3] = (int)((unsigned)DMODEL << 16);            // tile_dim0 = 128
    g1[4] = 32;                                       // tile_dim1 = 32
    g1[5] = DMODEL;                                   // tensor_dim0_stride = 128
    g1[6] = 0; g1[7] = 0;
    g2 = int32x4_t{0, 0, 0, 0};
    g3 = int32x4_t{0, 0, 0, 0};
#if __clang_major__ >= 23
    int32x8_t gpad = int32x8_t{0, 0, 0, 0, 0, 0, 0, 0};
    __builtin_amdgcn_tensor_load_to_lds(g0, g1, g2, g3, gpad, 0);
#else
    __builtin_amdgcn_tensor_load_to_lds(g0, g1, g2, g3, 0);
#endif
    __builtin_amdgcn_s_wait_tensorcnt(0);
  }
#else
  for (int i = tid; i < 32 * DMODEL; i += 256)
    hb[(i >> 7) * HLDA + (i & 127)] = hbg[(size_t)(tok0 + (i >> 7)) * DMODEL + (i & 127)];
#endif
  __syncthreads();

  const int wave = tid >> 5, lane = tid & 31, half = lane >> 4, r = lane & 15;
  const int mrow  = (wave & 1) * 16;
  const int ncol0 = (wave >> 1) * 16;

  v8f acc[8] = {};
  const unsigned short* pa = hb + (mrow + r) * HLDA + half * 8;
#pragma unroll
  for (int kb = 0; kb < DMODEL; kb += 32) {
    v16bf a = ld_bf16x16(pa + kb, pa + kb + 16);
#pragma unroll
    for (int i = 0; i < 8; ++i) {
      const int ng = v0 + ncol0 + i * 64 + r;
      v16bf b;
      if (ng < VOCAB_N) {
        const unsigned short* pb = Eb + (size_t)ng * DMODEL + half * 8 + kb;
        b = ld_bf16x16(pb, pb + 16);
      } else {
        b = bf_zero();
      }
      acc[i] = __builtin_amdgcn_wmma_f32_16x16x32_bf16(false, a, false, b, (short)0, acc[i], false, false);
    }
  }
#pragma unroll
  for (int i = 0; i < 8; ++i) {
    const int ng = v0 + ncol0 + i * 64 + r;
    if (ng < VOCAB_N) {
#pragma unroll
      for (int j = 0; j < 8; ++j) {
        const int m = tok0 + mrow + half * 8 + j;
        out[(size_t)m * VOCAB_N + ng] = acc[i][j];
      }
    }
  }
}

// ---------------- launcher ----------------
extern "C" void kernel_launch(void* const* d_in, const int* in_sizes, int n_in,
                              void* d_out, int out_size, void* d_ws, size_t ws_size,
                              hipStream_t stream) {
  (void)in_sizes; (void)n_in; (void)out_size; (void)ws_size;
  const int*   x  = (const int*)  d_in[0];
  const float* E  = (const float*)d_in[1];
  const float* P  = (const float*)d_in[2];
  const float* W0 = (const float*)d_in[3];
  const float* b0 = (const float*)d_in[4];
  const float* W1 = (const float*)d_in[5];
  const float* b1 = (const float*)d_in[6];
  const float* W2 = (const float*)d_in[7];
  const float* b2 = (const float*)d_in[8];
  const float* W3 = (const float*)d_in[9];
  const float* rm = (const float*)d_in[11];
  const float* rg = (const float*)d_in[12];

  char* ws = (char*)d_ws;
  float*          h   = (float*)(ws + OFF_H);
  float*          v   = (float*)(ws + OFF_V);
  float*          xi  = (float*)(ws + OFF_XI);
  unsigned short* W0B = (unsigned short*)(ws + OFF_W0B);
  unsigned short* W0T = (unsigned short*)(ws + OFF_W0T);
  unsigned short* W1B = (unsigned short*)(ws + OFF_W1B);
  unsigned short* W1T = (unsigned short*)(ws + OFF_W1T);
  unsigned short* W2B = (unsigned short*)(ws + OFF_W2B);
  unsigned short* W2T = (unsigned short*)(ws + OFF_W2T);
  unsigned short* Eb  = (unsigned short*)(ws + OFF_EB);
  unsigned short* hbg = (unsigned short*)(ws + OFF_HB);

  conv_w_kernel<<<(256 * 512 + 255) / 256, 256, 0, stream>>>(W0, W0B, W0T, 256, 512);
  conv_w_kernel<<<(512 * 512 + 255) / 256, 256, 0, stream>>>(W1, W1B, W1T, 512, 512);
  conv_w_kernel<<<(512 * 512 + 255) / 256, 256, 0, stream>>>(W2, W2B, W2T, 512, 512);
  conv_e_kernel<<<(VOCAB_N * DMODEL + 255) / 256, 256, 0, stream>>>(E, Eb, VOCAB_N * DMODEL);

  embed_kernel<<<(NTOK * DMODEL) / 256, 256, 0, stream>>>(x, E, P, h, v);

  for (int l = 0; l < NLAYER; ++l) {
    cumsum_kernel<<<8, DMODEL, 0, stream>>>(h, xi);
    sarf_mlp_fused<<<NTOK / 32, 256, MLP_LDS_BYTES, stream>>>(
        xi, h, v, hbg, W0T, W0B, W1T, W1B, W2T, W2B, b0, b1, b2, W3, rm, rg);
  }

  dim3 lgrid(NTOK / 32, (VOCAB_N + 511) / 512);  // 128 x 99
  sarf_logits<<<lgrid, 256, 0, stream>>>(hbg, Eb, (float*)d_out);
}
